// NanoGPTBlock_10788957848092
// MI455X (gfx1250) — matrix operand order repaired and checked by reading, not compiled, not run on hardware
//
#include <hip/hip_runtime.h>

// ---------------------------------------------------------------------------
// NanoGPT block for gfx1250 (MI455X). bf16 WMMA (v_wmma_f32_16x16x32_bf16)
// for all matmuls, f32 accumulation, f32 softmax/LayerNorm.
// GEMM K-loop is double-buffered through LDS (one barrier per iteration,
// global loads for tile k+1 in flight while tile k feeds the WMMA pipe).
// ---------------------------------------------------------------------------

typedef __attribute__((ext_vector_type(16))) __bf16 v16bf;
typedef __attribute__((ext_vector_type(8)))  __bf16 v8bf;
typedef __attribute__((ext_vector_type(8)))  float  v8f;

#define D_MODEL 1024
#define BATCH   4
#define SEQ     2048
#define BT      (BATCH * SEQ)   // 8192

static __device__ inline __bf16 f2bf(float f) {
    union { float f; unsigned u; } in; in.f = f;
    unsigned u = in.u;
    u += 0x7FFFu + ((u >> 16) & 1u);           // round-to-nearest-even
    union { unsigned short s; __bf16 h; } out;
    out.s = (unsigned short)(u >> 16);
    return out.h;
}

static __device__ inline v16bf cat16(v8bf lo, v8bf hi) {
    return __builtin_shufflevector(lo, hi, 0,1,2,3,4,5,6,7,8,9,10,11,12,13,14,15);
}

// ---------------------------------------------------------------------------
// Weight prep: z=0..2 -> transpose+convert w_q/w_k/w_v to bf16 (Bt layout),
//              z=3..4 -> straight convert W1/W2 (used as y @ W.T, so Bt == W).
// ---------------------------------------------------------------------------
__global__ void prep_weights(const float* __restrict__ wq, const float* __restrict__ wk,
                             const float* __restrict__ wv, const float* __restrict__ W1,
                             const float* __restrict__ W2,
                             __bf16* __restrict__ wqT, __bf16* __restrict__ wkT,
                             __bf16* __restrict__ wvT, __bf16* __restrict__ w1b,
                             __bf16* __restrict__ w2b) {
    int idx = blockIdx.x * 256 + threadIdx.x;      // 0 .. D*D-1
    int z = blockIdx.y;
    int r = idx >> 10;          // / D_MODEL
    int c = idx & 1023;         // % D_MODEL
    const float* src = (z == 0) ? wq : (z == 1) ? wk : (z == 2) ? wv : (z == 3) ? W1 : W2;
    float v = src[idx];
    if (z < 3) {
        __bf16* dst = (z == 0) ? wqT : (z == 1) ? wkT : wvT;
        dst[(long)c * D_MODEL + r] = f2bf(v);      // transposed
    } else {
        ((z == 3) ? w1b : w2b)[idx] = f2bf(v);
    }
}

// ---------------------------------------------------------------------------
// LayerNorm (faithful to reference): y = (x - mu/sqrt(var)) * g + b,
// var unbiased (ddof=1). One wave32 per row, 8 rows per block.
// ---------------------------------------------------------------------------
__global__ void ln_rows(const float* __restrict__ X, const float* __restrict__ g,
                        const float* __restrict__ b, __bf16* __restrict__ Y) {
    int row  = blockIdx.x * 8 + (threadIdx.x >> 5);
    int lane = threadIdx.x & 31;
    const float* x = X + (long)row * D_MODEL;
    float s = 0.f;
    for (int j = lane; j < D_MODEL; j += 32) s += x[j];
    #pragma unroll
    for (int o = 16; o > 0; o >>= 1) s += __shfl_xor(s, o, 32);
    float mu = s * (1.0f / D_MODEL);
    float vs = 0.f;
    for (int j = lane; j < D_MODEL; j += 32) { float d = x[j] - mu; vs += d * d; }
    #pragma unroll
    for (int o = 16; o > 0; o >>= 1) vs += __shfl_xor(vs, o, 32);
    float var = vs * (1.0f / (D_MODEL - 1));
    float c = mu * rsqrtf(var);                    // mu / sqrt(var)
    __bf16* y = Y + (long)row * D_MODEL;
    for (int j = lane; j < D_MODEL; j += 32)
        y[j] = f2bf((x[j] - c) * g[j] + b[j]);
}

// ---------------------------------------------------------------------------
// Causal softmax over one row t: reads S[j] for j<=t, writes bf16 probs,
// zeros above the diagonal (ws is poisoned, so we must write the full row).
// ---------------------------------------------------------------------------
__global__ void softmax_causal(const float* __restrict__ S, __bf16* __restrict__ P) {
    int row = blockIdx.x;                 // 0 .. BT-1
    int bb = row / SEQ, t = row - bb * SEQ;
    const float* s = S + (long)bb * SEQ * SEQ + (long)t * SEQ;
    __bf16*      p = P + (long)bb * SEQ * SEQ + (long)t * SEQ;
    int L = t + 1;
    __shared__ float red[8];
    int tid = threadIdx.x, lane = tid & 31, wv = tid >> 5;

    float mx = -3.4e38f;
    for (int j = tid; j < L; j += 256) mx = fmaxf(mx, s[j]);
    #pragma unroll
    for (int o = 16; o > 0; o >>= 1) mx = fmaxf(mx, __shfl_xor(mx, o, 32));
    if (lane == 0) red[wv] = mx;
    __syncthreads();
    float rowMax = red[0];
    #pragma unroll
    for (int k = 1; k < 8; ++k) rowMax = fmaxf(rowMax, red[k]);
    __syncthreads();

    float sum = 0.f;
    for (int j = tid; j < L; j += 256) sum += __expf(s[j] - rowMax);
    #pragma unroll
    for (int o = 16; o > 0; o >>= 1) sum += __shfl_xor(sum, o, 32);
    if (lane == 0) red[wv] = sum;
    __syncthreads();
    float tot = 0.f;
    #pragma unroll
    for (int k = 0; k < 8; ++k) tot += red[k];
    float inv = 1.0f / tot;

    for (int j = tid; j < SEQ; j += 256)
        p[j] = (j < L) ? f2bf(__expf(s[j] - rowMax) * inv) : f2bf(0.f);
}

// ---------------------------------------------------------------------------
// Generic bf16 WMMA GEMM: C(MxN) = A(MxK) * Bt(NxK)^T, f32 accumulate.
// 128x128 block tile, BLK_K=32, 8 waves; each wave owns a 64x32 tile
// (4x2 WMMA 16x16 accumulators). LDS tiles padded to 40 bf16/row (80B):
// 16 lanes * 80B covers all 64 LDS banks -> conflict-free ds_load_b128.
// Double-buffered LDS: one barrier/iter; next tile's global loads issue
// right after the barrier and land in LDS after the WMMA block, while
// global_prefetch_b8 warms the tile after next.
// Epilogues: 0 f32 store (*alpha), 1 bf16 store, 2 bf16 transposed store
// (per-batch V^T), 3 +bias,ReLU,bf16, 4 +bias,+residual,f32.
// ---------------------------------------------------------------------------
enum { EPI_F32 = 0, EPI_BF = 1, EPI_BF_T = 2, EPI_BIAS_RELU_BF = 3, EPI_BIAS_RES_F32 = 4 };

template <int EPI>
__global__ __launch_bounds__(256)
void gemm_bf16(const __bf16* __restrict__ A, int lda, long strideAb,
               const __bf16* __restrict__ Bt, int ldb, long strideBb,
               void* __restrict__ Cout, int ldc, long strideCb,
               const float* __restrict__ bias, const float* __restrict__ residual,
               float alpha, int tDim, int causal, int Kdim) {
    const int bz = blockIdx.z;
    A  += (long)bz * strideAb;
    Bt += (long)bz * strideBb;
    const int m0 = blockIdx.y * 128;
    const int n0 = blockIdx.x * 128;
    if (causal && n0 > m0 + 127) return;   // tile fully above causal diagonal

    __shared__ __align__(16) __bf16 As[2][128][40];
    __shared__ __align__(16) __bf16 Bs[2][128][40];

    const int tid   = threadIdx.x;
    const int lane  = tid & 31;
    const int wv    = tid >> 5;      // 0..7
    const int waveM = wv >> 2;       // 0..1  (64 rows each)
    const int waveN = wv & 3;        // 0..3  (32 cols each)

    v8f acc[4][2];
    const v8f vzero = {0.f, 0.f, 0.f, 0.f, 0.f, 0.f, 0.f, 0.f};
    #pragma unroll
    for (int i = 0; i < 4; ++i)
        #pragma unroll
        for (int j = 0; j < 2; ++j) acc[i][j] = vzero;

    // --- per-thread global->LDS copy coords: 256 thr * 4 * v8bf = both tiles
    const int lr = tid >> 2;          // 0..63
    const int lc = (tid & 3) * 8;     // 0,8,16,24
    const __bf16* gA0 = A  + (long)(m0 + lr)      * lda + lc;
    const __bf16* gA1 = A  + (long)(m0 + lr + 64) * lda + lc;
    const __bf16* gB0 = Bt + (long)(n0 + lr)      * ldb + lc;
    const __bf16* gB1 = Bt + (long)(n0 + lr + 64) * ldb + lc;

    // --- per-wave fragment coords
    const int rowA  = waveM * 64 + (lane & 15);
    const int kselA = (lane >> 4) * 8;      // A frag: K {ks..ks+7, ks+16..ks+23}
    const int rowB  = waveN * 32 + (lane & 15);
    const int kselB = (lane >> 4) * 8;      // B frag: K {ks..ks+7, ks+16..ks+23}

    // --- prologue: tile 0 into buffer 0
    v8bf ra0 = *(const v8bf*)(gA0);
    v8bf ra1 = *(const v8bf*)(gA1);
    v8bf rb0 = *(const v8bf*)(gB0);
    v8bf rb1 = *(const v8bf*)(gB1);
    *(v8bf*)&As[0][lr][lc]      = ra0;
    *(v8bf*)&As[0][lr + 64][lc] = ra1;
    *(v8bf*)&Bs[0][lr][lc]      = rb0;
    *(v8bf*)&Bs[0][lr + 64][lc] = rb1;

    int cur = 0;
    for (int k0 = 0; k0 < Kdim; k0 += 32) {
        __syncthreads();               // buffer `cur` ready; buffer `cur^1` free
        const bool hasNext = (k0 + 32) < Kdim;
        if (hasNext) {                 // launch next tile's loads early
            ra0 = *(const v8bf*)(gA0 + k0 + 32);
            ra1 = *(const v8bf*)(gA1 + k0 + 32);
            rb0 = *(const v8bf*)(gB0 + k0 + 32);
            rb1 = *(const v8bf*)(gB1 + k0 + 32);
            if (k0 + 64 < Kdim) {      // warm tile after next (global_prefetch_b8)
                __builtin_prefetch(gA0 + k0 + 64, 0, 1);
                __builtin_prefetch(gB0 + k0 + 64, 0, 1);
            }
        }

        v16bf aF[4], bF[2];
        #pragma unroll
        for (int i = 0; i < 4; ++i) {
            const __bf16* p = &As[cur][rowA + i * 16][kselA];
            aF[i] = cat16(*(const v8bf*)p, *(const v8bf*)(p + 16));
        }
        #pragma unroll
        for (int j = 0; j < 2; ++j) {
            const __bf16* p = &Bs[cur][rowB + j * 16][kselB];
            bF[j] = cat16(*(const v8bf*)p, *(const v8bf*)(p + 16));
        }
        #pragma unroll
        for (int i = 0; i < 4; ++i)
            #pragma unroll
            for (int j = 0; j < 2; ++j)
                acc[i][j] = __builtin_amdgcn_wmma_f32_16x16x32_bf16(
                    false, aF[i], false, bF[j], (short)0, acc[i][j], false, false);

        if (hasNext) {                 // stash next tile while WMMAs drain
            *(v8bf*)&As[cur ^ 1][lr][lc]      = ra0;
            *(v8bf*)&As[cur ^ 1][lr + 64][lc] = ra1;
            *(v8bf*)&Bs[cur ^ 1][lr][lc]      = rb0;
            *(v8bf*)&Bs[cur ^ 1][lr + 64][lc] = rb1;
        }
        cur ^= 1;
    }

    // ---- epilogue: C-frag element r -> M = base+r+8*(lane>>4), N = lane&15
    const long cOff = (long)bz * strideCb;
    #pragma unroll
    for (int i = 0; i < 4; ++i) {
        #pragma unroll
        for (int j = 0; j < 2; ++j) {
            int mBase = m0 + waveM * 64 + i * 16 + 8 * (lane >> 4);
            int n     = n0 + waveN * 32 + j * 16 + (lane & 15);
            #pragma unroll
            for (int r = 0; r < 8; ++r) {
                int m = mBase + r;
                float v = acc[i][j][r] * alpha;
                if (EPI == EPI_F32) {
                    ((float*)Cout)[cOff + (long)m * ldc + n] = v;
                } else if (EPI == EPI_BF) {
                    ((__bf16*)Cout)[cOff + (long)m * ldc + n] = f2bf(v);
                } else if (EPI == EPI_BF_T) {
                    int bb = m / tDim, t = m - bb * tDim;
                    long nTot = (long)gridDim.x * 128;
                    ((__bf16*)Cout)[(long)bb * nTot * tDim + (long)n * tDim + t] = f2bf(v);
                } else if (EPI == EPI_BIAS_RELU_BF) {
                    v += bias[n];
                    ((__bf16*)Cout)[cOff + (long)m * ldc + n] = f2bf(fmaxf(v, 0.f));
                } else { // EPI_BIAS_RES_F32
                    v += bias[n] + residual[(long)m * ldc + n];
                    ((float*)Cout)[cOff + (long)m * ldc + n] = v;
                }
            }
        }
    }
}

// ---------------------------------------------------------------------------
// Host orchestration
// ---------------------------------------------------------------------------
extern "C" void kernel_launch(void* const* d_in, const int* in_sizes, int n_in,
                              void* d_out, int out_size, void* d_ws, size_t ws_size,
                              hipStream_t stream) {
    const float* x     = (const float*)d_in[0];
    const float* w_q   = (const float*)d_in[1];
    const float* w_k   = (const float*)d_in[2];
    const float* w_v   = (const float*)d_in[3];
    const float* ln1_g = (const float*)d_in[4];
    const float* ln1_b = (const float*)d_in[5];
    const float* ln2_g = (const float*)d_in[6];
    const float* ln2_b = (const float*)d_in[7];
    const float* W1    = (const float*)d_in[8];
    const float* b1    = (const float*)d_in[9];
    const float* W2    = (const float*)d_in[10];
    const float* b2    = (const float*)d_in[11];

    // --- workspace carve (256B aligned); dead buffers are aliased ---
    size_t off = 0;
    auto carve = [&](size_t bytes) -> char* {
        off = (off + 255) & ~(size_t)255;
        char* p = (char*)d_ws + off;
        off += bytes;
        return p;
    };
    __bf16* y1bf = (__bf16*)carve((size_t)BT * D_MODEL * 2);       // LN1 out
    __bf16* wqT  = (__bf16*)carve((size_t)D_MODEL * D_MODEL * 2);
    __bf16* wkT  = (__bf16*)carve((size_t)D_MODEL * D_MODEL * 2);
    __bf16* wvT  = (__bf16*)carve((size_t)D_MODEL * D_MODEL * 2);
    __bf16* w1b  = (__bf16*)carve((size_t)D_MODEL * D_MODEL * 2);
    __bf16* w2b  = (__bf16*)carve((size_t)D_MODEL * D_MODEL * 2);
    __bf16* Qbf  = (__bf16*)carve((size_t)BT * D_MODEL * 2);
    __bf16* Kbf  = (__bf16*)carve((size_t)BT * D_MODEL * 2);
    __bf16* Vt   = (__bf16*)carve((size_t)BATCH * D_MODEL * SEQ * 2);
    __bf16* P    = (__bf16*)carve((size_t)BATCH * SEQ * SEQ * 2);  // probs
    float*  S    = (float*) carve((size_t)BATCH * SEQ * SEQ * 4);  // scores
    // Aliases over dead buffers:
    float*  Y2   = (float*)S;     // attention out f32 (S dead after softmax)
    __bf16* Y2n  = Qbf;           // LN2 out         (Q dead after scores)
    __bf16* H    = Kbf;           // MLP hidden      (K dead after scores)

    const float gamma = 0.03125f;   // 1/sqrt(1024)

    // 1) weights -> bf16 (Q/K/V transposed)
    prep_weights<<<dim3((D_MODEL * D_MODEL) / 256, 5), 256, 0, stream>>>(
        w_q, w_k, w_v, W1, W2, wqT, wkT, wvT, w1b, w2b);

    // 2) LN1
    ln_rows<<<BT / 8, 256, 0, stream>>>(x, ln1_g, ln1_b, y1bf);

    // 3) Q, K, V projections (V stored transposed per batch: Vt[b][d][t])
    dim3 gProj(D_MODEL / 128, BT / 128, 1);
    gemm_bf16<EPI_BF><<<gProj, 256, 0, stream>>>(y1bf, D_MODEL, 0, wqT, D_MODEL, 0,
        Qbf, D_MODEL, 0, nullptr, nullptr, 1.f, 0, 0, D_MODEL);
    gemm_bf16<EPI_BF><<<gProj, 256, 0, stream>>>(y1bf, D_MODEL, 0, wkT, D_MODEL, 0,
        Kbf, D_MODEL, 0, nullptr, nullptr, 1.f, 0, 0, D_MODEL);
    gemm_bf16<EPI_BF_T><<<gProj, 256, 0, stream>>>(y1bf, D_MODEL, 0, wvT, D_MODEL, 0,
        Vt, D_MODEL, 0, nullptr, nullptr, 1.f, SEQ, 0, D_MODEL);

    // 4) scores = gamma * Q K^T per batch (causal tile skip)
    dim3 gScore(SEQ / 128, SEQ / 128, BATCH);
    gemm_bf16<EPI_F32><<<gScore, 256, 0, stream>>>(Qbf, D_MODEL, (long)SEQ * D_MODEL,
        Kbf, D_MODEL, (long)SEQ * D_MODEL, S, SEQ, (long)SEQ * SEQ,
        nullptr, nullptr, gamma, 0, 1, D_MODEL);

    // 5) causal softmax -> bf16 probs
    softmax_causal<<<BT, 256, 0, stream>>>(S, P);

    // 6) Y2 = P @ V per batch (B-matrix is Vt)
    dim3 gAV(D_MODEL / 128, SEQ / 128, BATCH);
    gemm_bf16<EPI_F32><<<gAV, 256, 0, stream>>>(P, SEQ, (long)SEQ * SEQ,
        Vt, SEQ, (long)D_MODEL * SEQ, Y2, D_MODEL, (long)SEQ * D_MODEL,
        nullptr, nullptr, 1.f, 0, 0, SEQ);

    // 7) LN2
    ln_rows<<<BT / 8, 256, 0, stream>>>(Y2, ln2_g, ln2_b, Y2n);

    // 8) MLP: H = relu(Y2n @ W1^T + b1)  (Bt == W1 as stored)
    gemm_bf16<EPI_BIAS_RELU_BF><<<gProj, 256, 0, stream>>>(Y2n, D_MODEL, 0,
        w1b, D_MODEL, 0, H, D_MODEL, 0, b1, nullptr, 1.f, 0, 0, D_MODEL);

    // 9) out = x + (H @ W2^T + b2)
    gemm_bf16<EPI_BIAS_RES_F32><<<gProj, 256, 0, stream>>>(H, D_MODEL, 0,
        w2b, D_MODEL, 0, d_out, D_MODEL, 0, b2, x, 1.f, 0, 0, D_MODEL);
}
